// MoEMultiBranchFFN_53446573031460
// MI455X (gfx1250) — compile-verified, matrix operand order checked by reading, not run on hardware
//
#include <hip/hip_runtime.h>
#include <math.h>

// ---------------- problem constants (from setup_inputs) ----------------
#define BB   64
#define TT   196
#define DD   512
#define FF   2048
#define EE   8
#define NBR  2
#define TE   512
#define NTOK (BB * TT)          // 12544
#define TILE_M 64
#define KF     32               // F-chunk == WMMA K for bf16
#define NTILES (NTOK / TILE_M)  // 196

typedef __bf16 bf16_t;
typedef __attribute__((ext_vector_type(16))) __bf16 v16bf;
typedef __attribute__((ext_vector_type(8)))  float  v8f;
typedef __attribute__((ext_vector_type(4)))  unsigned v4u;
typedef __attribute__((ext_vector_type(8)))  unsigned v8u;

// ---------------- helpers ----------------
__device__ inline float gelu_tanh(float v) {
  const float c = 0.7978845608028654f;
  float u = c * (v + 0.044715f * v * v * v);
  return 0.5f * v * (1.0f + tanhf(u));
}
__device__ inline float silu(float v) { return v / (1.0f + __expf(-v)); }

__device__ inline v8f wmma_bf16(v16bf a, v16bf b, v8f c) {
  // D = A(16x32 bf16) * B(32x16 bf16) + C(16x16 f32)
  return __builtin_amdgcn_wmma_f32_16x16x32_bf16(
      /*neg_a=*/false, a, /*neg_b=*/false, b,
      /*c_mod=*/(short)0, c, /*reuse_a=*/false, /*reuse_b=*/false);
}

// CDNA5 async global->LDS 16B copy (tracked on ASYNCcnt).
__device__ inline void g2l_b128(void* lds_dst, const void* gsrc) {
  unsigned loff = (unsigned)(size_t)lds_dst;  // low 32 bits of generic shared ptr = LDS byte offset
  asm volatile("global_load_async_to_lds_b128 %0, %1, off"
               :: "v"(loff), "v"(gsrc)
               : "memory");
}
__device__ inline void wait_async0() {
  asm volatile("s_wait_asynccnt 0x0" ::: "memory");
}

// CDNA5 Tensor Data Mover: one instruction DMAs a 2D tile (16-bit elements,
// tile_d0 contiguous elems per row, tile_d1 rows, row stride stride_d0 elems)
// from global memory into LDS (packed contiguously). D# per CDNA5 ISA.
// Issue from ONE wave (EXEC ignored; TENSORcnt is per-wave).
__device__ inline void tdm_load_2d(void* lds_dst, const void* gsrc,
                                   unsigned tile_d0, unsigned tile_d1,
                                   unsigned tensor_d0, unsigned tensor_d1,
                                   unsigned long long stride_d0_elems) {
  unsigned long long ga = (unsigned long long)(size_t)gsrc;
  v4u g0;
  g0[0] = 1u;                                   // count=1 (valid user descriptor)
  g0[1] = (unsigned)(size_t)lds_dst;            // lds_addr (bytes)
  g0[2] = (unsigned)ga;                         // global_addr[31:0]
  g0[3] = (unsigned)((ga >> 32) & 0x01FFFFFFull) | (2u << 30);  // addr[56:32] | type=2
  v8u g1;
  g1[0] = 1u << 16;                             // data_size=1 (2 bytes); mask=0; no pad/iter
  g1[1] = (tensor_d0 & 0xFFFFu) << 16;          // tensor_dim0[15:0]  @bits[63:48]
  g1[2] = (tensor_d0 >> 16) | ((tensor_d1 & 0xFFFFu) << 16);  // dim0 hi | dim1 lo
  g1[3] = (tensor_d1 >> 16) | (tile_d0 << 16);  // dim1 hi | tile_dim0
  g1[4] = tile_d1 & 0xFFFFu;                    // tile_dim1 ; tile_dim2 = 0 (2D)
  g1[5] = (unsigned)stride_d0_elems;            // tensor_dim0_stride[31:0]
  g1[6] = (unsigned)((stride_d0_elems >> 32) & 0xFFFFull);  // stride[47:32]; dim1_stride=0
  g1[7] = 0u;
  asm volatile("tensor_load_to_lds %0, %1"
               :: "s"(g0), "s"(g1)
               : "memory");
}

// Fragment load from an LDS tile where the "lane-major" dim (A: row m, B: col n)
// has stride ld_elems and k is contiguous. Per WMMA bf16 layout:
//   element j -> k = 16*(j>>3) + 8*(lane>>4) + (j&7)
// i.e. two contiguous 16-byte runs -> 2x ds_load_b128.
__device__ inline v16bf frag_ld(const bf16_t* base, int ld_elems, int lane) {
  int r = lane & 15, hs = (lane >> 4) & 1;
  const bf16_t* p = base + r * ld_elems + hs * 8;
  union { uint4 q[2]; v16bf v; } u;
  u.q[0] = *(const uint4*)(p);       // k = hs*8 + 0..7
  u.q[1] = *(const uint4*)(p + 16);  // k = 16 + hs*8 + 0..7
  return u.v;
}

// ---------------- K1: fp32 -> bf16 convert WITH transpose ----------------
// src: [exp][K][N] fp32  ->  dst: [exp][N][K] bf16   (exp = blockIdx.y)
__global__ __launch_bounds__(256) void k_cvt_t(const float* __restrict__ src,
                                               bf16_t* __restrict__ dst,
                                               int K, int N) {
  size_t sb = (size_t)blockIdx.y * K * N;
  int units = N * (K >> 3);
  for (int u = blockIdx.x * blockDim.x + threadIdx.x; u < units;
       u += gridDim.x * blockDim.x) {
    int kb = (u / N) * 8;  // 8 consecutive k per thread
    int n = u % N;         // consecutive threads -> consecutive n (coalesced reads)
    union { bf16_t h[8]; uint4 q; } t;
#pragma unroll
    for (int i = 0; i < 8; ++i)
      t.h[i] = (bf16_t)src[sb + (size_t)(kb + i) * N + n];
    *(uint4*)(dst + sb + (size_t)n * K + kb) = t.q;
  }
}

// ---------------- K2: LayerNorm + gate softmax/argmax + token lists --------
__global__ __launch_bounds__(256) void k_gate(
    const float* __restrict__ x, const float* __restrict__ ln_g,
    const float* __restrict__ ln_b, const float* __restrict__ gate_w,
    const float* __restrict__ gate_b, bf16_t* __restrict__ hn,
    float* __restrict__ gp, int* __restrict__ counts, int* __restrict__ order) {
  __shared__ float sh[DD];
  __shared__ float red[256];
  __shared__ float lg[EE];

  int t = blockIdx.x;
  int tid = threadIdx.x;
  float a0 = x[(size_t)t * DD + tid];
  float a1 = x[(size_t)t * DD + tid + 256];

  red[tid] = a0 + a1;
  __syncthreads();
  for (int s = 128; s > 0; s >>= 1) {
    if (tid < s) red[tid] += red[tid + s];
    __syncthreads();
  }
  float mu = red[0] * (1.0f / DD);
  __syncthreads();
  red[tid] = a0 * a0 + a1 * a1;
  __syncthreads();
  for (int s = 128; s > 0; s >>= 1) {
    if (tid < s) red[tid] += red[tid + s];
    __syncthreads();
  }
  float var = red[0] * (1.0f / DD) - mu * mu;
  float rs = rsqrtf(var + 1e-5f);
  __syncthreads();

  for (int br = 0; br < NBR; ++br) {
    float h0 = (a0 - mu) * rs * ln_g[br * DD + tid] + ln_b[br * DD + tid];
    float h1 = (a1 - mu) * rs * ln_g[br * DD + tid + 256] + ln_b[br * DD + tid + 256];
    sh[tid] = h0;
    sh[tid + 256] = h1;
    size_t hb = ((size_t)(br * NTOK + t)) * DD;
    hn[hb + tid] = (bf16_t)h0;
    hn[hb + tid + 256] = (bf16_t)h1;
    __syncthreads();

    // wave w computes logit for expert w (wave32: 8 waves == 8 experts)
    int e = tid >> 5, l = tid & 31;
    float sum = 0.0f;
    for (int i = l; i < DD; i += 32)
      sum += sh[i] * gate_w[((size_t)br * DD + i) * EE + e];
    for (int off = 16; off > 0; off >>= 1) sum += __shfl_down(sum, off);
    if (l == 0) lg[e] = sum + gate_b[br * EE + e];
    __syncthreads();

    if (tid == 0) {
      float m = lg[0];
      int am = 0;
      for (int i = 1; i < EE; ++i)
        if (lg[i] > m) { m = lg[i]; am = i; }
      float den = 0.0f;
      for (int i = 0; i < EE; ++i) den += __expf(lg[i] - m);
      gp[br * NTOK + t] = 1.0f / den;  // top-1 softmax prob
      int slot = atomicAdd(&counts[br * EE + am], 1);
      order[(size_t)(br * EE + am) * NTOK + slot] = t;
    }
    __syncthreads();
  }
}

// ---------------- K3: fused expert FFN with WMMA bf16 ----------------
// One block = 64 gathered tokens x full 512 output cols for one (branch, expert).
// Token gather: global_load_async_to_lds_b128 (ASYNCcnt, per-lane gather).
// Weight tiles: TDM tensor_load_to_lds (TENSORcnt, one DMA per 2D tile).
// All WMMA fragment loads: 2x ds_load_b128 from n-major LDS tiles.
__global__ __launch_bounds__(256) void k_expert(
    const bf16_t* __restrict__ hn, const bf16_t* __restrict__ w1t,
    const bf16_t* __restrict__ w2t, const float* __restrict__ b1,
    const float* __restrict__ b2, const int* __restrict__ counts,
    const int* __restrict__ order, const float* __restrict__ gp,
    float* __restrict__ moe) {
  __shared__ __align__(16) bf16_t sHn[TILE_M * DD];   // 64 KB (CDNA5: 320 KB LDS/WGP)
  __shared__ __align__(16) bf16_t sW[KF * DD];        // 32 KB (W1 chunk [32n x 512k] | W2 chunk [512n x 32k])
  __shared__ __align__(16) bf16_t sAct[TILE_M * KF];  // 4 KB
  __shared__ int sTok[TILE_M];
  __shared__ float sP[TILE_M];

  int bid = blockIdx.x;
  int br = bid / (EE * NTILES);
  int rem = bid % (EE * NTILES);
  int e = rem / NTILES;
  int tile = rem % NTILES;
  int be = br * EE + e;
  int cnt = counts[be];
  int row0 = tile * TILE_M;
  if (row0 >= cnt) return;
  int nrows = cnt - row0;
  if (nrows > TILE_M) nrows = TILE_M;

  int tid = threadIdx.x;
  int lane = tid & 31, wave = tid >> 5;
  int mt = wave >> 1;   // 0..3 : 16-row m-tile
  int half = wave & 1;  // 0..1 : half of the N range
  int hs = (lane >> 4) & 1;
  int nloc = lane & 15;

  if (tid < TILE_M) {
    int g = row0 + tid;
    if (g >= cnt) g = cnt - 1;  // pad rows duplicate last token; stores guarded
    int tok = order[(size_t)be * NTOK + g];
    sTok[tid] = tok;
    sP[tid] = gp[br * NTOK + tok];
  }
  __syncthreads();

  // async gather Hn tile [64 x 512] bf16 (4 threads/row, 16B per copy)
  {
    int r = tid >> 2, sub = tid & 3;
    int tok = sTok[r];
    const uint4* src = (const uint4*)(hn + ((size_t)(br * NTOK + tok)) * DD) + sub * 16;
    uint4* dst = (uint4*)(sHn + r * DD) + sub * 16;
#pragma unroll
    for (int j = 0; j < 16; ++j) g2l_b128(dst + j, src + j);
  }
  wait_async0();  // each wave drains its own ASYNCcnt before the barrier

  const bf16_t* w1te = w1t + (size_t)be * FF * DD;  // [F][D] n-major
  const bf16_t* w2te = w2t + (size_t)be * DD * FF;  // [D][F] n-major
  const float* b1e = b1 + (size_t)be * FF;
  const float* b2e = b2 + (size_t)be * DD;

  v8f oacc[16];
#pragma unroll
  for (int j = 0; j < 16; ++j)
#pragma unroll
    for (int r = 0; r < 8; ++r) oacc[j][r] = 0.0f;

  for (int fc = 0; fc < FF; fc += KF) {
    // phase 1: TDM-stage W1 chunk: 32 rows x 512 elems, contiguous rows
    if (fc + KF < FF)  // CDNA5 global_prefetch_b8 on next chunk
      __builtin_prefetch(w1te + (size_t)(fc + KF + (tid & 31)) * DD, 0, 1);
    if (wave == 0) {
      tdm_load_2d(sW, w1te + (size_t)fc * DD,
                  /*tile_d0=*/DD, /*tile_d1=*/KF,
                  /*tensor_d0=*/DD, /*tensor_d1=*/FF,
                  /*stride_d0=*/DD);
      __builtin_amdgcn_s_wait_tensorcnt(0);
    }
    __syncthreads();

    // phase 2: ACT[64x32] = gelu(Hn * W1c + b1); each wave one 16x16 tile
    {
      v8f acc;
#pragma unroll
      for (int r = 0; r < 8; ++r) acc[r] = 0.0f;
#pragma unroll
      for (int kk = 0; kk < DD / 32; ++kk) {
        v16bf a = frag_ld(sHn + (mt * 16) * DD + kk * 32, DD, lane);
        v16bf b = frag_ld(sW + (half * 16) * DD + kk * 32, DD, lane);
        acc = wmma_bf16(a, b, acc);
      }
      float bias = b1e[fc + half * 16 + nloc];
#pragma unroll
      for (int r = 0; r < 8; ++r) {
        int m = mt * 16 + r + 8 * hs;  // C/D layout: m = r + 8*(lane>>4)
        float v = gelu_tanh(acc[r] + bias);
        sAct[m * KF + half * 16 + nloc] = (bf16_t)v;
      }
    }
    __syncthreads();

    // phase 3: TDM-stage W2 chunk: 512 rows x 32 elems, row stride FF (strided 2D)
    if (wave == 0) {
      tdm_load_2d(sW, w2te + fc,
                  /*tile_d0=*/KF, /*tile_d1=*/DD,
                  /*tensor_d0=*/FF, /*tensor_d1=*/DD,
                  /*stride_d0=*/FF);
      __builtin_amdgcn_s_wait_tensorcnt(0);
    }
    __syncthreads();

    // phase 4: OUT[64x512] += ACT * W2c ; 16 n-tiles per wave, A reused,
    // B fragment software-pipelined one tile ahead to overlap dscnt waits.
    {
      v16bf a = frag_ld(sAct + (mt * 16) * KF, KF, lane);
      v16bf bcur = frag_ld(sW + (half * 256) * KF, KF, lane);
#pragma unroll
      for (int j = 0; j < 16; ++j) {
        v16bf bnext = bcur;
        if (j < 15)
          bnext = frag_ld(sW + (half * 256 + (j + 1) * 16) * KF, KF, lane);
        oacc[j] = wmma_bf16(a, bcur, oacc[j]);
        bcur = bnext;
      }
    }
    __syncthreads();
  }

  // scatter: moe[br][tok][col] = p * (out + b2)   (each token covered exactly once)
#pragma unroll
  for (int j = 0; j < 16; ++j) {
    int col = half * 256 + j * 16 + nloc;
    float bias = b2e[col];
#pragma unroll
    for (int r = 0; r < 8; ++r) {
      int m = mt * 16 + r + 8 * hs;
      if (m < nrows) {
        int tok = sTok[m];
        moe[((size_t)(br * NTOK + tok)) * DD + col] = (oacc[j][r] + bias) * sP[m];
      }
    }
  }
}

// ---------------- K4a: eo = silu(emb) @ emb_w + emb_b  [64 x 1024] -------
__global__ __launch_bounds__(256) void k_emb(const float* __restrict__ emb,
                                             const float* __restrict__ emb_w,
                                             const float* __restrict__ emb_b,
                                             float* __restrict__ eo) {
  __shared__ float se[TE];
  int b = blockIdx.x, tid = threadIdx.x;
  for (int i = tid; i < TE; i += 256) se[i] = silu(emb[(size_t)b * TE + i]);
  __syncthreads();
  float acc[4];
#pragma unroll
  for (int jj = 0; jj < 4; ++jj) acc[jj] = emb_b[tid + jj * 256];
  for (int k = 0; k < TE; ++k) {
    float s = se[k];
#pragma unroll
    for (int jj = 0; jj < 4; ++jj)
      acc[jj] += s * emb_w[(size_t)k * (2 * DD) + tid + jj * 256];
  }
#pragma unroll
  for (int jj = 0; jj < 4; ++jj)
    eo[(size_t)b * (2 * DD) + tid + jj * 256] = acc[jj];
}

// ---------------- K4b: FiLM + SiLU + out-proj + residual ----------------
__global__ __launch_bounds__(256) void k_final(
    const float* __restrict__ x, const float* __restrict__ moe,
    const float* __restrict__ eo, const float* __restrict__ sn_g,
    const float* __restrict__ sn_b, const float* __restrict__ out_w,
    const float* __restrict__ out_b, float* __restrict__ out) {
  __shared__ float ss[DD];
  __shared__ float red[256];
  int t = blockIdx.x, tid = threadIdx.x;
  int b = t / TT;

  float m0 = 0.5f * (moe[(size_t)t * DD + tid] + moe[(size_t)(NTOK + t) * DD + tid]);
  float m1 = 0.5f * (moe[(size_t)t * DD + tid + 256] + moe[(size_t)(NTOK + t) * DD + tid + 256]);

  red[tid] = m0 + m1;
  __syncthreads();
  for (int s = 128; s > 0; s >>= 1) {
    if (tid < s) red[tid] += red[tid + s];
    __syncthreads();
  }
  float mu = red[0] * (1.0f / DD);
  __syncthreads();
  red[tid] = m0 * m0 + m1 * m1;
  __syncthreads();
  for (int s = 128; s > 0; s >>= 1) {
    if (tid < s) red[tid] += red[tid + s];
    __syncthreads();
  }
  float var = red[0] * (1.0f / DD) - mu * mu;
  float rs = rsqrtf(var + 1e-5f);

  {
    const float* eob = eo + (size_t)b * (2 * DD);
    float h = (m0 - mu) * rs * sn_g[tid] + sn_b[tid];
    h = h * (1.0f + eob[tid]) + eob[DD + tid];
    ss[tid] = silu(h);
    float h2 = (m1 - mu) * rs * sn_g[tid + 256] + sn_b[tid + 256];
    h2 = h2 * (1.0f + eob[tid + 256]) + eob[DD + tid + 256];
    ss[tid + 256] = silu(h2);
  }
  __syncthreads();

  float acc0 = out_b[tid], acc1 = out_b[tid + 256];
  for (int k = 0; k < DD; ++k) {
    float s = ss[k];
    acc0 += s * out_w[(size_t)k * DD + tid];
    acc1 += s * out_w[(size_t)k * DD + tid + 256];
  }
  out[(size_t)t * DD + tid] = x[(size_t)t * DD + tid] + acc0;
  out[(size_t)t * DD + tid + 256] = x[(size_t)t * DD + tid + 256] + acc1;
}

// ---------------- host launcher ----------------
extern "C" void kernel_launch(void* const* d_in, const int* in_sizes, int n_in,
                              void* d_out, int out_size, void* d_ws, size_t ws_size,
                              hipStream_t stream) {
  const float* x      = (const float*)d_in[0];
  const float* emb    = (const float*)d_in[1];
  const float* ln_g   = (const float*)d_in[2];
  const float* ln_b   = (const float*)d_in[3];
  const float* gate_w = (const float*)d_in[4];
  const float* gate_b = (const float*)d_in[5];
  const float* w1     = (const float*)d_in[6];
  const float* b1     = (const float*)d_in[7];
  const float* w2     = (const float*)d_in[8];
  const float* b2     = (const float*)d_in[9];
  const float* emb_w  = (const float*)d_in[10];
  const float* emb_b  = (const float*)d_in[11];
  const float* sn_g   = (const float*)d_in[12];
  const float* sn_b   = (const float*)d_in[13];
  const float* out_w  = (const float*)d_in[14];
  const float* out_b  = (const float*)d_in[15];
  float* out = (float*)d_out;

  char* base = (char*)d_ws;
  size_t off = 0;
  auto alloc = [&](size_t bytes) -> void* {
    void* p = base + off;
    off += (bytes + 255) & ~(size_t)255;
    return p;
  };
  bf16_t* hn   = (bf16_t*)alloc((size_t)NBR * NTOK * DD * sizeof(bf16_t));    // 25.7 MB
  bf16_t* w1t  = (bf16_t*)alloc((size_t)NBR * EE * DD * FF * sizeof(bf16_t)); // 33.6 MB [be][F][D]
  bf16_t* w2t  = (bf16_t*)alloc((size_t)NBR * EE * FF * DD * sizeof(bf16_t)); // 33.6 MB [be][D][F]
  float* moe   = (float*)alloc((size_t)NBR * NTOK * DD * sizeof(float));      // 51.4 MB
  float* gp    = (float*)alloc((size_t)NBR * NTOK * sizeof(float));
  int* counts  = (int*)alloc((size_t)NBR * EE * sizeof(int));
  int* order   = (int*)alloc((size_t)NBR * EE * NTOK * sizeof(int));
  float* eo    = (float*)alloc((size_t)BB * 2 * DD * sizeof(float));

  hipMemsetAsync(counts, 0, NBR * EE * sizeof(int), stream);

  // convert + transpose weights to bf16 n-major (one-time, HBM-bound)
  k_cvt_t<<<dim3(512, NBR * EE), 256, 0, stream>>>(w1, w1t, DD, FF);  // [D][F] -> [F][D]
  k_cvt_t<<<dim3(512, NBR * EE), 256, 0, stream>>>(w2, w2t, FF, DD);  // [F][D] -> [D][F]

  k_gate<<<NTOK, 256, 0, stream>>>(x, ln_g, ln_b, gate_w, gate_b, hn, gp, counts, order);

  k_expert<<<NBR * EE * NTILES, 256, 0, stream>>>(hn, w1t, w2t, b1, b2, counts,
                                                  order, gp, moe);

  k_emb<<<BB, 256, 0, stream>>>(emb, emb_w, emb_b, eo);
  k_final<<<NTOK, 256, 0, stream>>>(x, moe, eo, sn_g, sn_b, out_w, out_b, out);
}
